// LearnedCrossStockAttention_9792525435558
// MI455X (gfx1250) — compile-verified, matrix operand order checked by reading, not run on hardware
//
#include <hip/hip_runtime.h>
#include <hip/hip_bf16.h>

#define HIDDEN  1024
#define NSTOCK  6000
#define HEADS   32
#define HD      64
#define TOPK    200
#define BATCH   4
#define SEQ     128

typedef __attribute__((ext_vector_type(16))) __bf16 v16bf;
typedef __attribute__((ext_vector_type(8)))  float  v8f;

union FragU { uint4 u[2]; v16bf v; };

__device__ __forceinline__ float gelu_exact(float x) {
    return 0.5f * x * (1.0f + erff(x * 0.70710678118654752f));
}

// 16-elem bf16 fragment per ISA 7.12.2 (16-bit A 16x32 / B 32x16):
// per lane: elems 0..7 -> K = 8*lh + 0..7, elems 8..15 -> K = 8*lh + 16..23.
__device__ __forceinline__ v16bf load_frag(const __bf16* __restrict__ base, int stride, int row, int lh) {
    const __bf16* p = base + (size_t)row * stride + 8 * lh;
    FragU f;
    f.u[0] = *(const uint4*)(p);
    f.u[1] = *(const uint4*)(p + 16);
    return f.v;
}

#define BM 128
#define BN 128
#define BK 32
#define ASTR (BK + 8)   // LDS row stride (bf16 elems): 80B rows spread across banks

// Async-stage one 128x32 bf16 tile (rows of 64B) from global into LDS rows of stride 80B.
// ASYNCcnt-tracked; caller must s_wait_asynccnt + barrier before consuming.
__device__ __forceinline__ void stage_async(const __bf16* __restrict__ gbase, int gstride,
                                            int rowlimit, unsigned lds_base, int tid)
{
#pragma unroll
    for (int i = 0; i < 2; i++) {
        const int linear = tid + 256 * i;   // 0..511 slots of 16B
        const int row = linear >> 2;
        const int q   = linear & 3;
        if (row < rowlimit) {
            const __bf16* gp = gbase + (size_t)row * gstride + q * 8;
            const unsigned la = lds_base + (unsigned)(row * (ASTR * 2) + q * 16);
            asm volatile("global_load_async_to_lds_b128 %0, %1, off"
                         :: "v"(la), "v"(gp) : "memory");
        }
    }
}

__device__ __forceinline__ void wait_async_all() {
    asm volatile("s_wait_asynccnt 0x0" ::: "memory");
}

// C = act(A[MxK] * Bt[NxK]^T + bias); A,Bt bf16 row-major; epilogue 0=none,1=bias,2=bias+gelu
// out_mode: 0 = f32, 1 = bf16, 2 = both. Requires K%32==0, N%128==0; M guarded.
__global__ __launch_bounds__(256) void gemm_bf16_kernel(
    const __bf16* __restrict__ A, const __bf16* __restrict__ Bt,
    const float* __restrict__ bias, float* __restrict__ Cf, __bf16* __restrict__ Cbf,
    int M, int N, int K, int epilogue, int out_mode)
{
    __shared__ __align__(16) __bf16 smA[2][BM * ASTR];
    __shared__ __align__(16) __bf16 smB[2][BM * ASTR];

    const int tid  = threadIdx.x;
    const int m0   = blockIdx.y * BM;
    const int n0   = blockIdx.x * BN;
    const int wave = tid >> 5, lane = tid & 31;
    const int wm   = wave >> 1, wn = wave & 1;       // 4x2 waves -> 32x64 per wave
    const int lh   = lane >> 4, l16 = lane & 15;

    const unsigned aB[2] = { (unsigned)(size_t)&smA[0][0], (unsigned)(size_t)&smA[1][0] };
    const unsigned bB[2] = { (unsigned)(size_t)&smB[0][0], (unsigned)(size_t)&smB[1][0] };

    int mlim = M - m0; if (mlim > BM) mlim = BM;

    v8f acc[2][4];
#pragma unroll
    for (int i = 0; i < 2; i++)
#pragma unroll
        for (int j = 0; j < 4; j++)
#pragma unroll
            for (int r = 0; r < 8; r++) acc[i][j][r] = 0.0f;

    // prologue: stage k-tile 0 into buffer 0
    stage_async(A  + (size_t)m0 * K, K, mlim, aB[0], tid);
    stage_async(Bt + (size_t)n0 * K, K, BM,   bB[0], tid);

    for (int k0 = 0; k0 < K; k0 += BK) {
        const int cb = (k0 >> 5) & 1;
        wait_async_all();        // own wave's async copies into buf cb are done
        __syncthreads();         // everyone's copies done; prior readers of cb^1 finished

        if (k0 + BK < K) {       // stage next tile into the other buffer (overlaps WMMA)
            stage_async(A  + (size_t)m0 * K + k0 + BK, K, mlim, aB[cb ^ 1], tid);
            stage_async(Bt + (size_t)n0 * K + k0 + BK, K, BM,   bB[cb ^ 1], tid);
        }

        const __bf16* Asp = smA[cb];
        const __bf16* Bsp = smB[cb];
        v16bf af[2], bfr[4];
#pragma unroll
        for (int ia = 0; ia < 2; ia++)
            af[ia] = load_frag(Asp, ASTR, wm * 32 + ia * 16 + l16, lh);
#pragma unroll
        for (int jb = 0; jb < 4; jb++)
            bfr[jb] = load_frag(Bsp, ASTR, wn * 64 + jb * 16 + l16, lh);
#pragma unroll
        for (int ia = 0; ia < 2; ia++)
#pragma unroll
            for (int jb = 0; jb < 4; jb++)
                acc[ia][jb] = __builtin_amdgcn_wmma_f32_16x16x32_bf16(
                    false, af[ia], false, bfr[jb], (short)0, acc[ia][jb], false, false);
    }

    // epilogue: C layout -> m = 8*lh + r, n = l16 within each 16x16 tile
#pragma unroll
    for (int ia = 0; ia < 2; ia++)
#pragma unroll
        for (int jb = 0; jb < 4; jb++)
#pragma unroll
            for (int r = 0; r < 8; r++) {
                const int m = m0 + wm * 32 + ia * 16 + lh * 8 + r;
                const int n = n0 + wn * 64 + jb * 16 + l16;
                if (m >= M) continue;
                float v = acc[ia][jb][r];
                if (epilogue >= 1) v += bias[n];
                if (epilogue == 2) v = gelu_exact(v);
                if (out_mode != 1) Cf [(size_t)m * N + n] = v;
                if (out_mode >= 1) Cbf[(size_t)m * N + n] = (__bf16)v;
            }
}

// scores[h][s][n] = gate[b][n][h] * (1/8) * sum_d Q[b,s,h,d] * K[n,h,d]
__global__ __launch_bounds__(256) void scores_kernel(
    const __bf16* __restrict__ Qbf, const __bf16* __restrict__ Kbf,
    const float* __restrict__ gates, float* __restrict__ scores, int b)
{
    const int h   = blockIdx.y;
    const int n0  = blockIdx.x * BN;
    const int tid = threadIdx.x;
    const int wave = tid >> 5, lane = tid & 31;
    const int wm = wave >> 1, wn = wave & 1;
    const int lh = lane >> 4, l16 = lane & 15;

    v8f acc[2][4];
#pragma unroll
    for (int i = 0; i < 2; i++)
#pragma unroll
        for (int j = 0; j < 4; j++)
#pragma unroll
            for (int r = 0; r < 8; r++) acc[i][j][r] = 0.0f;

    const __bf16* qbase = Qbf + (size_t)b * SEQ * HIDDEN + h * HD;

#pragma unroll
    for (int k0 = 0; k0 < HD; k0 += 32) {
        v16bf af[2], bfr[4];
#pragma unroll
        for (int ia = 0; ia < 2; ia++) {
            const int s = wm * 32 + ia * 16 + l16;
            const __bf16* p = qbase + (size_t)s * HIDDEN + k0 + 8 * lh;
            FragU f; f.u[0] = *(const uint4*)p; f.u[1] = *(const uint4*)(p + 16);
            af[ia] = f.v;
        }
#pragma unroll
        for (int jb = 0; jb < 4; jb++) {
            int n = n0 + wn * 64 + jb * 16 + l16;
            if (n >= NSTOCK) n = NSTOCK - 1;          // clamp; stores guarded below
            const __bf16* p = Kbf + (size_t)n * HIDDEN + h * HD + k0 + 8 * lh;
            FragU f; f.u[0] = *(const uint4*)p; f.u[1] = *(const uint4*)(p + 16);
            bfr[jb] = f.v;
        }
#pragma unroll
        for (int ia = 0; ia < 2; ia++)
#pragma unroll
            for (int jb = 0; jb < 4; jb++)
                acc[ia][jb] = __builtin_amdgcn_wmma_f32_16x16x32_bf16(
                    false, af[ia], false, bfr[jb], (short)0, acc[ia][jb], false, false);
    }

#pragma unroll
    for (int ia = 0; ia < 2; ia++)
#pragma unroll
        for (int jb = 0; jb < 4; jb++)
#pragma unroll
            for (int r = 0; r < 8; r++) {
                const int s = wm * 32 + ia * 16 + lh * 8 + r;
                const int n = n0 + wn * 64 + jb * 16 + l16;
                if (n < NSTOCK) {
                    const float g = gates[((size_t)b * NSTOCK + n) * HEADS + h];
                    scores[((size_t)h * SEQ + s) * NSTOCK + n] = acc[ia][jb][r] * 0.125f * g;
                }
            }
}

// elementwise f32 -> bf16 (count multiple of 4)
__global__ __launch_bounds__(256) void cvt_bf16_kernel(
    const float* __restrict__ s, __bf16* __restrict__ d, int n4)
{
    const int i = blockIdx.x * 256 + threadIdx.x;
    if (i < n4) {
        float4 v = ((const float4*)s)[i];
        union { __bf16 h[4]; uint2 u; } o;
        o.h[0] = (__bf16)v.x; o.h[1] = (__bf16)v.y; o.h[2] = (__bf16)v.z; o.h[3] = (__bf16)v.w;
        ((uint2*)d)[i] = o.u;
    }
}

// src[K][N] f32 -> dst[N][K] bf16 (K,N multiples of 32)
__global__ __launch_bounds__(256) void transpose_bf16_kernel(
    const float* __restrict__ src, __bf16* __restrict__ dst, int K, int N)
{
    __shared__ float tile[32][33];
    const int k0 = blockIdx.y * 32, n0 = blockIdx.x * 32;
    const int tx = threadIdx.x & 31, tg = threadIdx.x >> 5;  // 8 row-groups
#pragma unroll
    for (int i = 0; i < 4; i++) {
        const int kr = tg + i * 8;
        tile[kr][tx] = src[(size_t)(k0 + kr) * N + n0 + tx];
    }
    __syncthreads();
#pragma unroll
    for (int i = 0; i < 4; i++) {
        const int nr = tg + i * 8;
        dst[(size_t)(n0 + nr) * K + k0 + tx] = (__bf16)tile[tx][nr];
    }
}

// query_combined[b] = mean_s(qf[b,s,:]) + KV[ids[b],:]  (bf16 out; consumed only as GEMM A)
__global__ __launch_bounds__(256) void qc_kernel(
    const float* __restrict__ qf, const int* __restrict__ ids,
    const float* __restrict__ KV, __bf16* __restrict__ qcb)
{
    const int b = blockIdx.x, tid = threadIdx.x;
    for (int j = tid; j < HIDDEN; j += 256) {
        float s = 0.f;
        for (int t = 0; t < SEQ; t++) s += qf[((size_t)b * SEQ + t) * HIDDEN + j];
        qcb[b * HIDDEN + j] = (__bf16)(s * (1.0f / SEQ) + KV[(size_t)ids[b] * HIDDEN + j]);
    }
}

// gates[b][n][h] = sigmoid( gelu(G0[n]+qcW1[b]+bg1) @ Wg2[:,h] + bg2[h] )
__global__ __launch_bounds__(256) void gates_kernel(
    const float* __restrict__ G0, const float* __restrict__ qcW1,
    const float* __restrict__ bg1, const float* __restrict__ Wg2,
    const float* __restrict__ bg2, float* __restrict__ gates)
{
    const int n = blockIdx.x, b = blockIdx.y, tid = threadIdx.x;
    float acch[HEADS];
#pragma unroll
    for (int h = 0; h < HEADS; h++) acch[h] = 0.f;

    const float* g0row = G0 + (size_t)n * HIDDEN;
    const float* qrow  = qcW1 + (size_t)b * HIDDEN;
    const int j0 = tid * 4;
    float4 g  = *(const float4*)(g0row + j0);
    float4 q  = *(const float4*)(qrow + j0);
    float4 bb = *(const float4*)(bg1 + j0);
    float hv[4] = { gelu_exact(g.x + q.x + bb.x), gelu_exact(g.y + q.y + bb.y),
                    gelu_exact(g.z + q.z + bb.z), gelu_exact(g.w + q.w + bb.w) };
#pragma unroll
    for (int jj = 0; jj < 4; jj++) {
        const float* wrow = Wg2 + (size_t)(j0 + jj) * HEADS;
        const float hj = hv[jj];
#pragma unroll
        for (int h = 0; h < HEADS; h++) acch[h] += hj * wrow[h];
    }
#pragma unroll
    for (int h = 0; h < HEADS; h++)
        for (int off = 16; off > 0; off >>= 1) acch[h] += __shfl_down(acch[h], off, 32);

    __shared__ float sg[8][HEADS];
    const int wave = tid >> 5, lane = tid & 31;
    if (lane == 0)
        for (int h = 0; h < HEADS; h++) sg[wave][h] = acch[h];
    __syncthreads();
    if (tid < HEADS) {
        float s = bg2[tid];
        for (int w = 0; w < 8; w++) s += sg[w][tid];
        gates[((size_t)b * NSTOCK + n) * HEADS + tid] = 1.0f / (1.0f + expf(-s));
    }
}

// Per (s,h): exact top-200 via radix threshold, bitonic sort desc, softmax, context.
__global__ __launch_bounds__(256) void topk_kernel(
    const float* __restrict__ scores,      // [H][S][N] for batch b
    const float* __restrict__ V,           // [N][HIDDEN]
    float* __restrict__ attn_out,          // [B][H][S][K]
    int*   __restrict__ attended_out,      // [B][H][K]
    __bf16* __restrict__ context,          // [B*S][HIDDEN] bf16 (GEMM A operand)
    int b)
{
    const int s = blockIdx.x, h = blockIdx.y, tid = threadIdx.x;
    const float* row = scores + ((size_t)h * SEQ + s) * NSTOCK;

    float    val[24];
    unsigned key[24];
#pragma unroll
    for (int i = 0; i < 24; i++) {
        const int n = tid + 256 * i;
        const bool ok = (n < NSTOCK);
        const float f = ok ? row[n] : 0.f;
        val[i] = f;
        unsigned u = __float_as_uint(f);
        u = (u & 0x80000000u) ? ~u : (u | 0x80000000u);
        key[i] = ok ? u : 0u;
    }

    __shared__ int sred[256];
    unsigned prefix = 0;
    for (int bit = 31; bit >= 0; bit--) {
        const unsigned X = prefix | (1u << bit);
        int c = 0;
#pragma unroll
        for (int i = 0; i < 24; i++) c += (key[i] >= X) ? 1 : 0;
        sred[tid] = c; __syncthreads();
        for (int off = 128; off > 0; off >>= 1) {
            if (tid < off) sred[tid] += sred[tid + off];
            __syncthreads();
        }
        const int total = sred[0]; __syncthreads();
        if (total >= TOPK) prefix = X;
    }
    const unsigned T = prefix;

    __shared__ float skey[256];
    __shared__ int   sidx[256];
    __shared__ int   ctr;
    if (tid == 0) ctr = 0;
    skey[tid] = -INFINITY; sidx[tid] = 0;
    __syncthreads();
#pragma unroll
    for (int i = 0; i < 24; i++)
        if (key[i] > T) {
            const int pos = atomicAdd(&ctr, 1);
            if (pos < TOPK) { skey[pos] = val[i]; sidx[pos] = tid + 256 * i; }
        }
    __syncthreads();
#pragma unroll
    for (int i = 0; i < 24; i++)
        if (key[i] == T) {
            const int pos = atomicAdd(&ctr, 1);
            if (pos < TOPK) { skey[pos] = val[i]; sidx[pos] = tid + 256 * i; }
        }
    __syncthreads();

    for (int k = 2; k <= 256; k <<= 1)
        for (int j = k >> 1; j > 0; j >>= 1) {
            const int ixj = tid ^ j;
            if (ixj > tid) {
                const bool desc = ((tid & k) == 0);
                const float a0 = skey[tid], a1 = skey[ixj];
                const bool sw = desc ? (a0 < a1) : (a0 > a1);
                if (sw) {
                    const int i0 = sidx[tid];
                    skey[tid] = a1; skey[ixj] = a0;
                    sidx[tid] = sidx[ixj]; sidx[ixj] = i0;
                }
            }
            __syncthreads();
        }

    __shared__ float sredf[256];
    __shared__ float sp[256];
    const float mx = skey[0];
    const float e = (tid < TOPK) ? expf(skey[tid] - mx) : 0.f;
    sredf[tid] = e; __syncthreads();
    for (int off = 128; off > 0; off >>= 1) {
        if (tid < off) sredf[tid] += sredf[tid + off];
        __syncthreads();
    }
    const float sum = sredf[0]; __syncthreads();
    const float p = e / sum;
    sp[tid] = p; __syncthreads();

    if (tid < TOPK) {
        attn_out[(((size_t)b * HEADS + h) * SEQ + s) * TOPK + tid] = p;
        if (s == SEQ - 1)
            attended_out[((size_t)b * HEADS + h) * TOPK + tid] = sidx[tid];
    }

    const int d = tid & 63, g = tid >> 6;
    float partial = 0.f;
    for (int k = g; k < TOPK; k += 4)
        partial += sp[k] * V[(size_t)sidx[k] * HIDDEN + h * HD + d];
    sredf[tid] = partial; __syncthreads();
    if (tid < 64) {
        const float c = sredf[tid] + sredf[tid + 64] + sredf[tid + 128] + sredf[tid + 192];
        context[((size_t)b * SEQ + s) * HIDDEN + h * HD + tid] = (__bf16)c;
    }
}

// output = LN(qf + out) * g + b
__global__ __launch_bounds__(256) void ln_kernel(
    const float* __restrict__ qf, const float* __restrict__ outb,
    const float* __restrict__ lng, const float* __restrict__ lnb,
    float* __restrict__ out)
{
    const int rowid = blockIdx.x;
    const int tid = threadIdx.x;
    const float* xq = qf   + (size_t)rowid * HIDDEN;
    const float* xo = outb + (size_t)rowid * HIDDEN;
    float x[4], lsum = 0.f, lsq = 0.f;
#pragma unroll
    for (int i = 0; i < 4; i++) {
        const int j = tid * 4 + i;
        const float v = xq[j] + xo[j];
        x[i] = v; lsum += v; lsq += v * v;
    }
    __shared__ float s1[256], s2[256];
    s1[tid] = lsum; s2[tid] = lsq; __syncthreads();
    for (int off = 128; off > 0; off >>= 1) {
        if (tid < off) { s1[tid] += s1[tid + off]; s2[tid] += s2[tid + off]; }
        __syncthreads();
    }
    const float mu  = s1[0] * (1.0f / HIDDEN);
    const float var = s2[0] * (1.0f / HIDDEN) - mu * mu;
    const float inv = rsqrtf(var + 1e-5f);
#pragma unroll
    for (int i = 0; i < 4; i++) {
        const int j = tid * 4 + i;
        out[(size_t)rowid * HIDDEN + j] = (x[i] - mu) * inv * lng[j] + lnb[j];
    }
}

extern "C" void kernel_launch(void* const* d_in, const int* in_sizes, int n_in,
                              void* d_out, int out_size, void* d_ws, size_t ws_size,
                              hipStream_t stream)
{
    const float* qf   = (const float*)d_in[0];
    const int*   ids  = (const int*)  d_in[1];
    const float* emb  = (const float*)d_in[2];
    const float* pW1  = (const float*)d_in[3];
    const float* pb1  = (const float*)d_in[4];
    const float* pW2  = (const float*)d_in[5];
    const float* pb2  = (const float*)d_in[6];
    const float* Wq   = (const float*)d_in[7];
    const float* bq   = (const float*)d_in[8];
    const float* Wk   = (const float*)d_in[9];
    const float* bk   = (const float*)d_in[10];
    const float* Wv   = (const float*)d_in[11];
    const float* bv   = (const float*)d_in[12];
    const float* Wo   = (const float*)d_in[13];
    const float* bo   = (const float*)d_in[14];
    const float* Wg1  = (const float*)d_in[15];
    const float* bg1  = (const float*)d_in[16];
    const float* Wg2  = (const float*)d_in[17];
    const float* bg2  = (const float*)d_in[18];
    const float* lng  = (const float*)d_in[19];
    const float* lnb  = (const float*)d_in[20];

    // deterministic workspace carve-out
    char* ws = (char*)d_ws;
    size_t off = 0;
    auto carve = [&](size_t bytes) -> char* {
        char* p = ws + off;
        off += (bytes + 255) & ~(size_t)255;
        return p;
    };
    __bf16* embbf = (__bf16*)carve((size_t)NSTOCK * 256 * 2);
    __bf16* qfbf  = (__bf16*)carve((size_t)BATCH * SEQ * HIDDEN * 2);
    __bf16* pW1t  = (__bf16*)carve((size_t)HIDDEN * 256 * 2);
    __bf16* pW2t  = (__bf16*)carve((size_t)HIDDEN * HIDDEN * 2);
    __bf16* Wqt   = (__bf16*)carve((size_t)HIDDEN * HIDDEN * 2);
    __bf16* Wkt   = (__bf16*)carve((size_t)HIDDEN * HIDDEN * 2);
    __bf16* Wvt   = (__bf16*)carve((size_t)HIDDEN * HIDDEN * 2);
    __bf16* Wot   = (__bf16*)carve((size_t)HIDDEN * HIDDEN * 2);
    __bf16* Wg1at = (__bf16*)carve((size_t)HIDDEN * HIDDEN * 2);
    __bf16* Wg1bt = (__bf16*)carve((size_t)HIDDEN * HIDDEN * 2);
    __bf16* T1bf  = (__bf16*)carve((size_t)NSTOCK * HIDDEN * 2);
    float*  KV    = (float*) carve((size_t)NSTOCK * HIDDEN * 4);
    __bf16* KVbf  = (__bf16*)carve((size_t)NSTOCK * HIDDEN * 2);
    float*  Vbuf  = (float*) carve((size_t)NSTOCK * HIDDEN * 4);
    float*  G0    = (float*) carve((size_t)NSTOCK * HIDDEN * 4);
    __bf16* Qbf   = (__bf16*)carve((size_t)BATCH * SEQ * HIDDEN * 2);
    __bf16* Kbf   = (__bf16*)carve((size_t)NSTOCK * HIDDEN * 2);
    __bf16* qcbf  = (__bf16*)carve((size_t)BATCH * HIDDEN * 2);
    float*  qcW1  = (float*) carve((size_t)BATCH * HIDDEN * 4);
    float*  gatesB= (float*) carve((size_t)BATCH * NSTOCK * HEADS * 4);
    float*  scB   = (float*) carve((size_t)HEADS * SEQ * NSTOCK * 4);  // reused per batch
    __bf16* ctxbf = (__bf16*)carve((size_t)BATCH * SEQ * HIDDEN * 2);
    float*  outb  = (float*) carve((size_t)BATCH * SEQ * HIDDEN * 4);

    float* out0     = (float*)d_out;                             // [4,128,1024]
    float* attn_out = out0 + (size_t)BATCH * SEQ * HIDDEN;       // [4,32,128,200]
    int*   att_idx  = (int*)(attn_out + (size_t)BATCH * HEADS * SEQ * TOPK); // [4,32,200]

    const dim3 blk(256);
    const int mt6000 = (NSTOCK + BM - 1) / BM;   // 47
    const int nt1024 = HIDDEN / BN;              // 8
    const size_t HH = (size_t)HIDDEN * HIDDEN;

    // --- one-time conversions: activations -> bf16, weights -> bf16 [N][K] ---
    cvt_bf16_kernel<<<dim3((NSTOCK * 256 / 4 + 255) / 256), blk, 0, stream>>>(emb, embbf, NSTOCK * 256 / 4);
    cvt_bf16_kernel<<<dim3((BATCH * SEQ * HIDDEN / 4 + 255) / 256), blk, 0, stream>>>(qf, qfbf, BATCH * SEQ * HIDDEN / 4);
    transpose_bf16_kernel<<<dim3(HIDDEN / 32, 256 / 32),  blk, 0, stream>>>(pW1, pW1t, 256, HIDDEN);
    transpose_bf16_kernel<<<dim3(HIDDEN / 32, HIDDEN / 32), blk, 0, stream>>>(pW2, pW2t, HIDDEN, HIDDEN);
    transpose_bf16_kernel<<<dim3(HIDDEN / 32, HIDDEN / 32), blk, 0, stream>>>(Wq, Wqt, HIDDEN, HIDDEN);
    transpose_bf16_kernel<<<dim3(HIDDEN / 32, HIDDEN / 32), blk, 0, stream>>>(Wk, Wkt, HIDDEN, HIDDEN);
    transpose_bf16_kernel<<<dim3(HIDDEN / 32, HIDDEN / 32), blk, 0, stream>>>(Wv, Wvt, HIDDEN, HIDDEN);
    transpose_bf16_kernel<<<dim3(HIDDEN / 32, HIDDEN / 32), blk, 0, stream>>>(Wo, Wot, HIDDEN, HIDDEN);
    transpose_bf16_kernel<<<dim3(HIDDEN / 32, HIDDEN / 32), blk, 0, stream>>>(Wg1, Wg1at, HIDDEN, HIDDEN);
    transpose_bf16_kernel<<<dim3(HIDDEN / 32, HIDDEN / 32), blk, 0, stream>>>(Wg1 + HH, Wg1bt, HIDDEN, HIDDEN);

    // --- embed_proj: T1 = gelu(emb @ pW1 + pb1); KV = T1 @ pW2 + pb2 (f32 + bf16) ---
    gemm_bf16_kernel<<<dim3(nt1024, mt6000), blk, 0, stream>>>(embbf, pW1t, pb1, nullptr, T1bf, NSTOCK, HIDDEN, 256, 2, 1);
    gemm_bf16_kernel<<<dim3(nt1024, mt6000), blk, 0, stream>>>(T1bf, pW2t, pb2, KV, KVbf, NSTOCK, HIDDEN, HIDDEN, 1, 2);

    // --- query_combined (uses KV[ids] == embed_proj(emb[ids])) ---
    qc_kernel<<<dim3(BATCH), blk, 0, stream>>>(qf, ids, KV, qcbf);

    // --- Q (bf16), K (bf16), V (f32) ---
    gemm_bf16_kernel<<<dim3(nt1024, (BATCH * SEQ) / BM), blk, 0, stream>>>(qfbf, Wqt, bq, nullptr, Qbf, BATCH * SEQ, HIDDEN, HIDDEN, 1, 1);
    gemm_bf16_kernel<<<dim3(nt1024, mt6000), blk, 0, stream>>>(KVbf, Wkt, bk, nullptr, Kbf, NSTOCK, HIDDEN, HIDDEN, 1, 1);
    gemm_bf16_kernel<<<dim3(nt1024, mt6000), blk, 0, stream>>>(KVbf, Wvt, bv, Vbuf, nullptr, NSTOCK, HIDDEN, HIDDEN, 1, 0);

    // --- gate path: rel_in @ Wg1 factored into batch-invariant + tiny per-batch ---
    gemm_bf16_kernel<<<dim3(nt1024, mt6000), blk, 0, stream>>>(KVbf, Wg1bt, nullptr, G0, nullptr, NSTOCK, HIDDEN, HIDDEN, 0, 0);
    gemm_bf16_kernel<<<dim3(nt1024, 1), blk, 0, stream>>>(qcbf, Wg1at, nullptr, qcW1, nullptr, BATCH, HIDDEN, HIDDEN, 0, 0);
    gates_kernel<<<dim3(NSTOCK, BATCH), blk, 0, stream>>>(G0, qcW1, bg1, Wg2, bg2, gatesB);

    // --- attention per batch (scores buffer reused, stream-ordered) ---
    for (int b = 0; b < BATCH; b++) {
        scores_kernel<<<dim3(mt6000, HEADS), blk, 0, stream>>>(Qbf, Kbf, gatesB, scB, b);
        topk_kernel<<<dim3(SEQ, HEADS), blk, 0, stream>>>(scB, Vbuf, attn_out, att_idx, ctxbf, b);
    }

    // --- output projection + residual LayerNorm ---
    gemm_bf16_kernel<<<dim3(nt1024, (BATCH * SEQ) / BM), blk, 0, stream>>>(ctxbf, Wot, bo, outb, nullptr, BATCH * SEQ, HIDDEN, HIDDEN, 1, 0);
    ln_kernel<<<dim3(BATCH * SEQ), blk, 0, stream>>>(qf, outb, lng, lnb, out0);
}
